// Block2ProductKeyMemory_85959475462659
// MI455X (gfx1250) — matrix operand order repaired and checked by reading, not compiled, not run on hardware
//
#include <hip/hip_runtime.h>

#define DMODEL 1024
#define DMEM   512
#define BT     2048
#define NHEAD  8
#define DH     64
#define NK     512
#define TOPK   16

typedef __attribute__((ext_vector_type(16))) __bf16 v16bf;
typedef __attribute__((ext_vector_type(8)))  __bf16 v8bf;
typedef __attribute__((ext_vector_type(8)))  float  v8f;

// ---------- wave32 reductions ----------
__device__ __forceinline__ float wave_sum_f(float v) {
#pragma unroll
    for (int o = 16; o; o >>= 1) v += __shfl_xor(v, o, 32);
    return v;
}
__device__ __forceinline__ float wave_max_f(float v) {
#pragma unroll
    for (int o = 16; o; o >>= 1) v = fmaxf(v, __shfl_xor(v, o, 32));
    return v;
}

__device__ __forceinline__ v8bf pack8(const float4 a, const float4 b) {
    v8bf r;
    r[0] = (__bf16)a.x; r[1] = (__bf16)a.y; r[2] = (__bf16)a.z; r[3] = (__bf16)a.w;
    r[4] = (__bf16)b.x; r[5] = (__bf16)b.y; r[6] = (__bf16)b.z; r[7] = (__bf16)b.w;
    return r;
}

// ---------- CDNA5 async global->LDS copy (ASYNCcnt path) ----------
// LDS offset = low 32 bits of the flat shared address (LDS aperture base has
// zero low bits, ISA 10.2). GV mode: 64-bit vaddr, saddr = off.
__device__ __forceinline__ void async_copy_b128(void* lds_dst, const void* gsrc) {
    const unsigned lds = (unsigned)(uintptr_t)lds_dst;
    const unsigned long long ga = (unsigned long long)(uintptr_t)gsrc;
    asm volatile("global_load_async_to_lds_b128 %0, %1, off" :: "v"(lds), "v"(ga) : "memory");
}
__device__ __forceinline__ void wait_async() {
    asm volatile("s_wait_asynccnt 0x0" ::: "memory");
}

// ---------- WMMA fragment loaders (ISA 7.12.2 layouts), 128-bit LDS --------
__device__ __forceinline__ v16bf frag_a(const __bf16* base, int lane, int ld) {
    const int r  = lane & 15;
    const int kh = (lane >> 4) << 3;
    const __bf16* p = base + r * ld + kh;
    const v8bf lo = *(const v8bf*)p;
    const v8bf hi = *(const v8bf*)(p + 16);
    v16bf a;
#pragma unroll
    for (int i = 0; i < 8; ++i) { a[i] = lo[i]; a[8 + i] = hi[i]; }
    return a;
}
__device__ __forceinline__ v16bf frag_b(const __bf16* base, int lane, int ld) {
    const __bf16* p = base + ((lane >> 4) << 3) * ld + (lane & 15);
    v16bf b;
#pragma unroll
    for (int i = 0; i < 8; ++i) { b[i] = p[i * ld]; b[8 + i] = p[(16 + i) * ld]; }
    return b;
}

// ========== P0: one-time transpose+convert weights -> bf16 K-major =========
// src: N x K (f32, row-major)   dst: K x N (bf16)
__global__ __launch_bounds__(256) void transpose_cvt_kernel(
    const float* __restrict__ src, __bf16* __restrict__ dst, int N, int K) {
    __shared__ float T[64][65];
    const int k0 = blockIdx.x * 64, n0 = blockIdx.y * 64;
    const int tid = threadIdx.x;
    const int r = tid >> 2;                          // n within tile
#pragma unroll
    for (int j = 0; j < 4; ++j) {
        const int c = (tid & 3) * 16 + j * 4;        // k within tile
        const float4 v = *(const float4*)(src + (size_t)(n0 + r) * K + k0 + c);
        T[c + 0][r] = v.x; T[c + 1][r] = v.y; T[c + 2][r] = v.z; T[c + 3][r] = v.w;
    }
    __syncthreads();
    const int kr = tid >> 2, nb = (tid & 3) * 16;
    v8bf o0, o1;
#pragma unroll
    for (int i = 0; i < 8; ++i) {
        o0[i] = (__bf16)T[kr][nb + i];
        o1[i] = (__bf16)T[kr][nb + 8 + i];
    }
    __bf16* op = dst + (size_t)(k0 + kr) * N + n0 + nb;
    *(v8bf*)op = o0; *(v8bf*)(op + 8) = o1;
}

// ===================== K0: normalize keys -> bf16, TRANSPOSED [h][dh][key] ==
__global__ __launch_bounds__(256) void key_norm_kernel(
    const float* __restrict__ kl, const float* __restrict__ kr,
    __bf16* __restrict__ klt, __bf16* __restrict__ krt) {
    const int row  = blockIdx.x * 8 + (threadIdx.x >> 5);  // 0 .. 2*H*NK-1
    const int lane = threadIdx.x & 31;
    const bool left = row < NHEAD * NK;                    // uniform per block
    const float* src = left ? kl : kr;
    __bf16* dstT     = left ? klt : krt;
    const int r = left ? row : row - NHEAD * NK;
    const int h = r >> 9, k = r & (NK - 1);
    const float2 v = *(const float2*)(src + (size_t)r * DH + lane * 2);
    float ss = wave_sum_f(v.x * v.x + v.y * v.y);
    const float inv = 1.0f / fmaxf(sqrtf(ss), 1e-12f);
    dstT[((size_t)(h * DH + lane * 2))     * NK + k] = (__bf16)(v.x * inv);
    dstT[((size_t)(h * DH + lane * 2 + 1)) * NK + k] = (__bf16)(v.y * inv);
}

// ===================== K1: q = query @ Wq^T, fused per-head L2 norm =========
// Double-buffered: async B copy + A register-prefetch for tile i+1 overlap
// the WMMA work of tile i; one barrier per K-step.
__global__ __launch_bounds__(256) void qproj_norm_kernel(
    const float* __restrict__ query, const __bf16* __restrict__ WqT,
    __bf16* __restrict__ qn) {
    __shared__ __bf16 As[2][64][40];  // row-major [m][k]
    __shared__ __bf16 Bs[2][32][72];  // K-major  [k][n], 144B row stride
    __shared__ float  Qs[64][65];
    __shared__ float  invs[64];
    const int m0 = blockIdx.x * 64, n0 = blockIdx.y * 64;   // n tile == one head
    const int tid = threadIdx.x, wave = tid >> 5, lane = tid & 31;
    const int wr = (wave & 3) << 4, wc = (wave >> 2) << 5;
    const int sr = tid >> 2, scb = (tid & 3) * 8;           // A staging coords
    const int bk = tid >> 3, bnb = (tid & 7) * 8;           // B staging coords
    v8f acc0 = {}, acc1 = {};
    // ---- prologue: stage tile 0 into buffer 0
    async_copy_b128(&Bs[0][bk][bnb], WqT + (size_t)bk * DMEM + n0 + bnb);
    {
        const float* qp = query + (size_t)(m0 + sr) * DMODEL + scb;
        *(v8bf*)&As[0][sr][scb] = pack8(*(const float4*)qp, *(const float4*)(qp + 4));
    }
    wait_async();
    __syncthreads();
    for (int i = 0; i < DMODEL / 32; ++i) {
        const int cur = i & 1, nxt = cur ^ 1;
        const int ktn = (i + 1) * 32;
        const bool more = ktn < DMODEL;
        float4 ga0, ga1;
        if (more) {  // issue next tile's traffic before the compute
            async_copy_b128(&Bs[nxt][bk][bnb], WqT + (size_t)(ktn + bk) * DMEM + n0 + bnb);
            const float* qp = query + (size_t)(m0 + sr) * DMODEL + ktn + scb;
            ga0 = *(const float4*)qp; ga1 = *(const float4*)(qp + 4);
        }
        const v16bf a  = frag_a(&As[cur][wr][0], lane, 40);
        const v16bf b0 = frag_b(&Bs[cur][0][wc], lane, 72);
        const v16bf b1 = frag_b(&Bs[cur][0][wc + 16], lane, 72);
        acc0 = __builtin_amdgcn_wmma_f32_16x16x32_bf16(false, a, false, b0, (short)0, acc0, false, false);
        acc1 = __builtin_amdgcn_wmma_f32_16x16x32_bf16(false, a, false, b1, (short)0, acc1, false, false);
        if (more) *(v8bf*)&As[nxt][sr][scb] = pack8(ga0, ga1);  // latency covered
        wait_async();
        __syncthreads();
    }
    {   // C layout: N = lane&15, M = vgpr + 8*(lane>=16)
        const int col = lane & 15, rb = wr + ((lane >> 4) << 3);
#pragma unroll
        for (int i = 0; i < 8; ++i) {
            Qs[rb + i][wc + col]      = acc0[i];
            Qs[rb + i][wc + 16 + col] = acc1[i];
        }
    }
    __syncthreads();
    if (tid < 64) {                    // per-head L2 norm over the 64 cols
        float ss = 0.f;
#pragma unroll
        for (int c = 0; c < 64; ++c) { const float v = Qs[tid][c]; ss += v * v; }
        invs[tid] = 1.0f / fmaxf(sqrtf(ss), 1e-12f);
    }
    __syncthreads();
    {   // packed b128 writes of q_n
        const int r = tid >> 2, cb = (tid & 3) * 16;
        const float inv = invs[r];
        v8bf o0, o1;
#pragma unroll
        for (int i = 0; i < 8; ++i) {
            o0[i] = (__bf16)(Qs[r][cb + i] * inv);
            o1[i] = (__bf16)(Qs[r][cb + 8 + i] * inv);
        }
        __bf16* op = qn + (size_t)(m0 + r) * DMEM + n0 + cb;
        *(v8bf*)op = o0;
        *(v8bf*)(op + 8) = o1;
    }
}

// ===================== K2: scores (WMMA) + top-16 both halves ===============
__global__ __launch_bounds__(256) void scores_topk_kernel(
    const __bf16* __restrict__ qn, const __bf16* __restrict__ klt,
    const __bf16* __restrict__ krt,
    float* __restrict__ ls, float* __restrict__ rs,
    int* __restrict__ li, int* __restrict__ ri) {
    __shared__ float Ssc[2][16][512];                  // 64 KB
    const int bt0 = blockIdx.x * 16, h = blockIdx.y;
    const int tid = threadIdx.x, wave = tid >> 5, lane = tid & 31;
    const int side = wave >> 2;                        // waves 0-3 left, 4-7 right
    const __bf16* keysT = side ? krt : klt;            // [h][dh][key]
    v8f acc[8] = {};
#pragma unroll
    for (int ks = 0; ks < 2; ++ks) {
        v16bf a;                                       // A = q_n rows (b128 x2)
        {
            const int m = bt0 + (lane & 15), kh = (lane >> 4) << 3;
            const __bf16* p = qn + (size_t)m * DMEM + h * DH + ks * 32 + kh;
            const v8bf lo = *(const v8bf*)p;
            const v8bf hi = *(const v8bf*)(p + 16);
#pragma unroll
            for (int i = 0; i < 8; ++i) { a[i] = lo[i]; a[8 + i] = hi[i]; }
        }
#pragma unroll
        for (int t = 0; t < 8; ++t) {                  // 8 N-tiles per wave
            const int n0 = ((wave & 3) * 8 + t) * 16;
            const __bf16* p = keysT + ((size_t)(h * DH + ks * 32 + ((lane >> 4) << 3))) * NK + n0 + (lane & 15);
            v16bf b;                                   // column n = lane&15, k in table order
#pragma unroll
            for (int i = 0; i < 8; ++i) { b[i] = p[(size_t)i * NK]; b[8 + i] = p[(size_t)(16 + i) * NK]; }
            acc[t] = __builtin_amdgcn_wmma_f32_16x16x32_bf16(false, a, false, b, (short)0, acc[t], false, false);
        }
    }
#pragma unroll
    for (int t = 0; t < 8; ++t) {
        const int n0 = ((wave & 3) * 8 + t) * 16;
        const int col = n0 + (lane & 15), rb = (lane >> 4) << 3;
#pragma unroll
        for (int i = 0; i < 8; ++i) Ssc[side][rb + i][col] = acc[t][i];
    }
    __syncthreads();
    // 32 top-k tasks (16 rows x 2 sides), 4 per wave; iterative wave argmax
    for (int q = 0; q < 4; ++q) {
        const int task = wave * 4 + q;
        const int s = task >> 4, row = task & 15;
        float* p = &Ssc[s][row][0];
        const int bt = bt0 + row;
        float* outS = (s ? rs : ls) + ((size_t)bt * NHEAD + h) * TOPK;
        int*   outI = (s ? ri : li) + ((size_t)bt * NHEAD + h) * TOPK;
        for (int it = 0; it < TOPK; ++it) {
            float best = -3.4e38f; int bi = 0x7fffffff;
#pragma unroll
            for (int j = 0; j < 16; ++j) {
                const float v = p[lane + 32 * j];
                if (v > best) { best = v; bi = lane + 32 * j; }
            }
#pragma unroll
            for (int o = 16; o; o >>= 1) {             // tie-break: lower index
                const float ov = __shfl_xor(best, o, 32);
                const int   oi = __shfl_xor(bi, o, 32);
                if (ov > best || (ov == best && oi < bi)) { best = ov; bi = oi; }
            }
            if (lane == 0) { outS[it] = best; outI[it] = bi; p[bi] = -3.4e38f; }
        }
    }
}

// ===================== K3: softmax(256) + value gather (memory-bound) =======
__global__ __launch_bounds__(256) void gather_kernel(
    const float* __restrict__ ls, const float* __restrict__ rs,
    const int* __restrict__ li, const int* __restrict__ ri,
    const float* __restrict__ values, __bf16* __restrict__ z) {
    __shared__ float sw[NHEAD][256];
    __shared__ int   sidx[NHEAD][256];
    const int bt = blockIdx.x;
    const int h = threadIdx.x >> 5, lane = threadIdx.x & 31;   // wave == head
    const float* lsp = ls + ((size_t)bt * NHEAD + h) * TOPK;
    const float* rsp = rs + ((size_t)bt * NHEAD + h) * TOPK;
    const int*   lip = li + ((size_t)bt * NHEAD + h) * TOPK;
    const int*   rip = ri + ((size_t)bt * NHEAD + h) * TOPK;
    float sc[8]; int sl[8];
    float m = -3.4e38f;
#pragma unroll
    for (int j = 0; j < 8; ++j) {
        const int c = lane * 8 + j, i = c >> 4, jr = c & 15;
        const float s = lsp[i] + rsp[jr];
        sc[j] = s;
        sl[j] = h * (NK * NK) + lip[i] * NK + rip[jr];
        m = fmaxf(m, s);
    }
    m = wave_max_f(m);
    float sum = 0.f;
#pragma unroll
    for (int j = 0; j < 8; ++j) { sc[j] = __expf(sc[j] - m); sum += sc[j]; }
    const float inv = 1.0f / wave_sum_f(sum);
#pragma unroll
    for (int j = 0; j < 8; ++j) { sw[h][lane * 8 + j] = sc[j] * inv; sidx[h][lane * 8 + j] = sl[j]; }
    // wave-private LDS slice; same-wave LDS ordering suffices
    float a0 = 0.f, a1 = 0.f;
    for (int c = 0; c < 256; ++c) {
        const float w = sw[h][c];
        const int slot = sidx[h][c];
        const int nslot = sidx[h][(c + 1) & 255];
        __builtin_prefetch(values + (size_t)nslot * DH + lane * 2, 0, 0);
        const float2 v = *(const float2*)(values + (size_t)slot * DH + lane * 2);
        a0 += w * v.x; a1 += w * v.y;
    }
    z[(size_t)bt * DMEM + h * DH + lane * 2]     = (__bf16)a0;
    z[(size_t)bt * DMEM + h * DH + lane * 2 + 1] = (__bf16)a1;
}

// ===================== K4: out = z @ Wout^T (WMMA, double-buffered async) ===
__global__ __launch_bounds__(256) void outproj_kernel(
    const __bf16* __restrict__ z, const __bf16* __restrict__ WoutT,
    float* __restrict__ out) {
    __shared__ __bf16 As[2][64][40];
    __shared__ __bf16 Bs[2][32][72];
    const int m0 = blockIdx.x * 64, n0 = blockIdx.y * 64;
    const int tid = threadIdx.x, wave = tid >> 5, lane = tid & 31;
    const int wr = (wave & 3) << 4, wc = (wave >> 2) << 5;
    const int sr = tid >> 2, scb = (tid & 3) * 8;
    const int bk = tid >> 3, bnb = (tid & 7) * 8;
    v8f acc0 = {}, acc1 = {};
    // ---- prologue: stage tile 0 (fully async)
    async_copy_b128(&As[0][sr][scb], z + (size_t)(m0 + sr) * DMEM + scb);
    async_copy_b128(&Bs[0][bk][bnb], WoutT + (size_t)bk * DMODEL + n0 + bnb);
    wait_async();
    __syncthreads();
    for (int i = 0; i < DMEM / 32; ++i) {
        const int cur = i & 1, nxt = cur ^ 1;
        const int ktn = (i + 1) * 32;
        const bool more = ktn < DMEM;
        if (more) {  // next tile's DMA overlaps this tile's WMMAs
            async_copy_b128(&As[nxt][sr][scb], z + (size_t)(m0 + sr) * DMEM + ktn + scb);
            async_copy_b128(&Bs[nxt][bk][bnb], WoutT + (size_t)(ktn + bk) * DMODEL + n0 + bnb);
        }
        const v16bf a  = frag_a(&As[cur][wr][0], lane, 40);
        const v16bf b0 = frag_b(&Bs[cur][0][wc], lane, 72);
        const v16bf b1 = frag_b(&Bs[cur][0][wc + 16], lane, 72);
        acc0 = __builtin_amdgcn_wmma_f32_16x16x32_bf16(false, a, false, b0, (short)0, acc0, false, false);
        acc1 = __builtin_amdgcn_wmma_f32_16x16x32_bf16(false, a, false, b1, (short)0, acc1, false, false);
        wait_async();
        __syncthreads();
    }
    const int col = lane & 15, rb = wr + ((lane >> 4) << 3);
#pragma unroll
    for (int i = 0; i < 8; ++i) {
        out[(size_t)(m0 + rb + i) * DMODEL + n0 + wc + col]      = acc0[i];
        out[(size_t)(m0 + rb + i) * DMODEL + n0 + wc + 16 + col] = acc1[i];
    }
}

// ===================== K5: in-place LayerNorm ===============================
__global__ __launch_bounds__(256) void layernorm_kernel(
    float* __restrict__ out, const float* __restrict__ gamma,
    const float* __restrict__ beta) {
    __shared__ float red[8];
    const int row = blockIdx.x, tid = threadIdx.x;
    const int wave = tid >> 5, lane = tid & 31;
    float x[4]; float s = 0.f;
#pragma unroll
    for (int i = 0; i < 4; ++i) { x[i] = out[(size_t)row * DMODEL + tid + i * 256]; s += x[i]; }
    s = wave_sum_f(s);
    if (lane == 0) red[wave] = s;
    __syncthreads();
    float tot = 0.f;
#pragma unroll
    for (int w = 0; w < 8; ++w) tot += red[w];
    const float mu = tot * (1.0f / DMODEL);
    __syncthreads();
    float vs = 0.f;
#pragma unroll
    for (int i = 0; i < 4; ++i) { const float d = x[i] - mu; vs += d * d; }
    vs = wave_sum_f(vs);
    if (lane == 0) red[wave] = vs;
    __syncthreads();
    float vtot = 0.f;
#pragma unroll
    for (int w = 0; w < 8; ++w) vtot += red[w];
    const float rstd = rsqrtf(vtot * (1.0f / DMODEL) + 1e-5f);
#pragma unroll
    for (int i = 0; i < 4; ++i) {
        const int c = tid + i * 256;
        out[(size_t)row * DMODEL + c] = (x[i] - mu) * rstd * gamma[c] + beta[c];
    }
}

// ===================== launch ===============================================
extern "C" void kernel_launch(void* const* d_in, const int* in_sizes, int n_in,
                              void* d_out, int out_size, void* d_ws, size_t ws_size,
                              hipStream_t stream) {
    const float* query  = (const float*)d_in[0];
    const float* Wq     = (const float*)d_in[1];
    const float* kl     = (const float*)d_in[2];
    const float* kr     = (const float*)d_in[3];
    const float* values = (const float*)d_in[4];
    const float* Wout   = (const float*)d_in[5];
    const float* gamma  = (const float*)d_in[6];
    const float* beta   = (const float*)d_in[7];
    float* out = (float*)d_out;

    char* ws = (char*)d_ws;
    __bf16* klt   = (__bf16*)(ws + 0);              // 512 KB  [h][dh][key]
    __bf16* krt   = (__bf16*)(ws + (512u << 10));   // 512 KB  [h][dh][key]
    __bf16* qn    = (__bf16*)(ws + (1u << 20));     // 2 MB
    __bf16* WqT   = (__bf16*)(ws + (3u << 20));     // 1 MB  [k=1024][n=512]
    __bf16* WoutT = (__bf16*)(ws + (4u << 20));     // 1 MB  [k=512][n=1024]
    float*  ls    = (float*)(ws + (5u << 20));      // 1 MB
    float*  rs    = (float*)(ws + (6u << 20));      // 1 MB
    int*    li    = (int*)(ws + (7u << 20));        // 1 MB
    int*    ri    = (int*)(ws + (8u << 20));        // 1 MB
    __bf16* z     = (__bf16*)(ws + (9u << 20));     // 2 MB   (total 11 MB)

    transpose_cvt_kernel<<<dim3(DMODEL / 64, DMEM / 64), 256, 0, stream>>>(Wq, WqT, DMEM, DMODEL);
    transpose_cvt_kernel<<<dim3(DMEM / 64, DMODEL / 64), 256, 0, stream>>>(Wout, WoutT, DMODEL, DMEM);
    key_norm_kernel<<<(2 * NHEAD * NK) / 8, 256, 0, stream>>>(kl, kr, klt, krt);
    qproj_norm_kernel<<<dim3(BT / 64, DMEM / 64), 256, 0, stream>>>(query, WqT, qn);
    scores_topk_kernel<<<dim3(BT / 16, NHEAD), 256, 0, stream>>>(qn, klt, krt, ls, rs, li, ri);
    gather_kernel<<<BT, 256, 0, stream>>>(ls, rs, li, ri, values, z);
    outproj_kernel<<<dim3(BT / 64, DMODEL / 64), 256, 0, stream>>>(z, WoutT, out);
    layernorm_kernel<<<BT, 256, 0, stream>>>(out, gamma, beta);
}
